// MultiHeadSelfAttention_3856880632001
// MI455X (gfx1250) — matrix operand order repaired and checked by reading, not compiled, not run on hardware
//
#include <hip/hip_runtime.h>

// ---------------------------------------------------------------------------
// MI455X (gfx1250) multi-head self-attention, f16 WMMA pipeline, fp32 accum.
// B=4, S=2048, D=768, H=12, dh=64.
// ---------------------------------------------------------------------------

typedef __attribute__((ext_vector_type(16))) _Float16 v16h;
typedef __attribute__((ext_vector_type(8)))  _Float16 v8h;
typedef __attribute__((ext_vector_type(2)))  _Float16 h2;
typedef __attribute__((ext_vector_type(8)))  float    v8f;

#define WMMA_F32_F16(A, B, C) \
    __builtin_amdgcn_wmma_f32_16x16x32_f16(false, (A), false, (B), (short)0, (C), false, false)

// ---- WMMA fragment loaders (contiguous K order, same in A and B) ------

// A operand, 16x32 f16 (M x K), row-major source with leading dim `ld`.
// lanes 0-15 : row = lane,    K = 0..15 contiguous
// lanes 16-31: row = lane-16, K = 16..31 contiguous
__device__ __forceinline__ v16h load_a16x32(const _Float16* base, int ld, int lane) {
    int row  = lane & 15;
    int koff = (lane & 16) ? 16 : 0;
    const _Float16* p = base + (size_t)row * ld + koff;
    v8h lo = *(const v8h*)(p);
    v8h hi = *(const v8h*)(p + 8);
    v16h a;
#pragma unroll
    for (int i = 0; i < 8; ++i) { a[i] = lo[i]; a[i + 8] = hi[i]; }
    return a;
}

// B operand, 32x16 f16 (K x N) where logical B[k][n] = Rows[n][k]
// (multiply by Rows^T; Rows is row-major [N x K], base points at Rows[n0][k0]).
// lane n in 0..15 : column n, K = 0..15 contiguous; lanes 16-31: K = 16..31.
__device__ __forceinline__ v16h load_bT16(const _Float16* row_base, int ld, int lane) {
    int n    = lane & 15;
    int koff = (lane & 16) ? 16 : 0;
    return *(const v16h*)(row_base + (size_t)n * ld + koff);
}

// ---- gfx1250 async global->LDS copy (ASYNCcnt-tracked) --------------------
// Each lane copies 16 bytes: LDS[ldsOff] = MEM[gaddr].
__device__ __forceinline__ void async_copy_b128(unsigned ldsOff, const void* gaddr) {
    asm volatile("global_load_async_to_lds_b128 %0, %1, off"
                 :: "v"(ldsOff), "v"(gaddr)
                 : "memory");
}
__device__ __forceinline__ void wait_asynccnt0() {
    asm volatile("s_wait_asynccnt 0x0" ::: "memory");
}

// ---- conversion -----------------------------------------------------------

__global__ void cvt_f32_to_f16_kernel(const float* __restrict__ src,
                                      _Float16* __restrict__ dst,
                                      int n, float scale) {
    int i = blockIdx.x * blockDim.x + threadIdx.x;
    if (i < n) dst[i] = (_Float16)(src[i] * scale);
}

// ---- QKV projection: [8192,768] @ W^T -> [B,H,S,64] f16 -------------------
// grid: (12, 64, 3), 128 threads (4 waves); wave tile = 32x64 (B frags reused).
__global__ void qkv_gemm_kernel(const _Float16* __restrict__ xh,
                                const _Float16* __restrict__ wh,   // [3][768*768]
                                _Float16* __restrict__ qkv) {      // [3][4*12*2048*64]
    const int lane = threadIdx.x & 31;
    const int wv   = threadIdx.x >> 5;
    const int z    = blockIdx.z;
    const _Float16* W   = wh  + (size_t)z * 768 * 768;
    _Float16*       out = qkv + (size_t)z * 8192 * 768;

    const int m0 = blockIdx.y * 128 + wv * 32;
    const int n0 = blockIdx.x * 64;

    v8f acc[2][4] = {};
    for (int k = 0; k < 768; k += 32) {
        v16h a0 = load_a16x32(xh + (size_t)m0 * 768 + k,        768, lane);
        v16h a1 = load_a16x32(xh + (size_t)(m0 + 16) * 768 + k, 768, lane);
#pragma unroll
        for (int j = 0; j < 4; ++j) {
            v16h b = load_bT16(W + (size_t)(n0 + j * 16) * 768 + k, 768, lane);
            acc[0][j] = WMMA_F32_F16(a0, b, acc[0][j]);
            acc[1][j] = WMMA_F32_F16(a1, b, acc[1][j]);
        }
    }

    const int n  = lane & 15;
    const int mo = (lane & 16) ? 8 : 0;
#pragma unroll
    for (int hrow = 0; hrow < 2; ++hrow)
#pragma unroll
        for (int j = 0; j < 4; ++j)
#pragma unroll
            for (int r = 0; r < 8; ++r) {
                int m  = m0 + hrow * 16 + mo + r;   // b*2048 + s
                int c  = n0 + j * 16 + n;           // h*64 + d
                int bI = m >> 11, s = m & 2047, h = c >> 6, d = c & 63;
                out[(((size_t)bI * 12 + h) * 2048 + s) * 64 + d] =
                    (_Float16)acc[hrow][j][r];
            }
}

// ---- flash attention with causal mask -------------------------------------
// grid: (S/64=32, B*H=48), 128 threads (4 waves). Block = (b,h, 64-query tile);
// wave w owns 16 query rows. K tiles double-buffered via async global->LDS DMA.
__global__ void flash_attn_kernel(const _Float16* __restrict__ qh,
                                  const _Float16* __restrict__ kh,
                                  const _Float16* __restrict__ vh,
                                  _Float16* __restrict__ aoh) {
    __shared__ _Float16 Klds[2][64 * 64];  // K tile row-major, double-buffered
    __shared__ _Float16 Vt[64 * 64];       // V tile transposed: [d][key]
    __shared__ _Float16 Pb[4][16 * 64];    // per-wave P scratch: [qrow][key]

    const int tid  = threadIdx.x;
    const int lane = tid & 31;
    const int w    = tid >> 5;
    const int q0   = blockIdx.x * 64;
    const int bh   = blockIdx.y;           // b*12 + h

    const _Float16* Q = qh + (size_t)bh * 2048 * 64;
    const _Float16* K = kh + (size_t)bh * 2048 * 64;
    const _Float16* V = vh + (size_t)bh * 2048 * 64;

    const int qrow = q0 + w * 16;
    const int n    = lane & 15;
    const int mo   = (lane & 16) ? 8 : 0;

    // Q fragments live in registers for the whole key loop (dh=64 -> 2 frags).
    v16h aq0 = load_a16x32(Q + (size_t)qrow * 64 + 0,  64, lane);
    v16h aq1 = load_a16x32(Q + (size_t)qrow * 64 + 32, 64, lane);

    v8f   o[4] = {};
    float mrow[8], lrow[8];
#pragma unroll
    for (int r = 0; r < 8; ++r) { mrow[r] = -3.0e38f; lrow[r] = 0.0f; }

    // Prologue: async-stage K tile 0 into buffer 0 (8KB = 128 thr * 4 * 16B).
    {
        unsigned kbase = (unsigned)(size_t)&Klds[0][0];
        const char* g  = (const char*)K;
#pragma unroll
        for (int i = 0; i < 4; ++i)
            async_copy_b128(kbase + tid * 16 + i * 2048, g + tid * 16 + i * 2048);
    }

    int buf = 0;
    for (int kt = 0; kt <= q0; kt += 64) {
        wait_asynccnt0();      // this wave's K copies for tile `kt` done
        __syncthreads();       // all waves' copies visible; Vt/Pb reads done

        // Kick off async copy of the next K tile into the other buffer.
        if (kt + 64 <= q0) {
            unsigned kbase = (unsigned)(size_t)&Klds[buf ^ 1][0];
            const char* g  = (const char*)(K + (size_t)(kt + 64) * 64);
#pragma unroll
            for (int i = 0; i < 4; ++i)
                async_copy_b128(kbase + tid * 16 + i * 2048, g + tid * 16 + i * 2048);
            __builtin_prefetch(V + (size_t)(kt + 64) * 64 + tid * 32, 0, 0);
        }

        // ---- stage V tile transposed into LDS (packed b32 stores) ---------
        {
            int p     = tid & 31;          // key pair index: keys 2p, 2p+1
            int dbase = (tid >> 5) * 16;   // 16 of 64 head dims
            const _Float16* s0 = V + (size_t)(kt + 2 * p) * 64 + dbase;
            v8h a0 = *(const v8h*)(s0);
            v8h a1 = *(const v8h*)(s0 + 8);
            v8h b0 = *(const v8h*)(s0 + 64);
            v8h b1 = *(const v8h*)(s0 + 72);
#pragma unroll
            for (int i = 0; i < 8; ++i) {
                *(h2*)&Vt[(dbase + i) * 64 + 2 * p]     = (h2){a0[i], b0[i]};
                *(h2*)&Vt[(dbase + 8 + i) * 64 + 2 * p] = (h2){a1[i], b1[i]};
            }
        }
        __syncthreads();

        // ---- scores S = Q K^T (pre-scaled by 1/8 via wq), K from LDS ------
        v8f st[4];
#pragma unroll
        for (int j = 0; j < 4; ++j) {
            v8f  s  = {};
            v16h b0 = load_bT16(&Klds[buf][(size_t)(j * 16) * 64 + 0],  64, lane);
            s = WMMA_F32_F16(aq0, b0, s);
            v16h b1 = load_bT16(&Klds[buf][(size_t)(j * 16) * 64 + 32], 64, lane);
            s = WMMA_F32_F16(aq1, b1, s);
            st[j] = s;
        }

        // ---- causal mask + online softmax update --------------------------
#pragma unroll
        for (int r = 0; r < 8; ++r) {
            int   qi = qrow + mo + r;
            float mx = -3.0e38f;
#pragma unroll
            for (int j = 0; j < 4; ++j) {
                int ki = kt + j * 16 + n;
                if (ki > qi) st[j][r] = -3.0e30f;
                mx = fmaxf(mx, st[j][r]);
            }
#pragma unroll
            for (int off = 1; off < 16; off <<= 1)
                mx = fmaxf(mx, __shfl_xor(mx, off, 32));
            float mnew = fmaxf(mrow[r], mx);
            float corr = __expf(mrow[r] - mnew);
            float rs   = 0.0f;
#pragma unroll
            for (int j = 0; j < 4; ++j) {
                float p = __expf(st[j][r] - mnew);
                st[j][r] = p;
                rs += p;
            }
#pragma unroll
            for (int off = 1; off < 16; off <<= 1)
                rs += __shfl_xor(rs, off, 32);
            lrow[r] = lrow[r] * corr + rs;
            mrow[r] = mnew;
#pragma unroll
            for (int j = 0; j < 4; ++j) o[j][r] *= corr;
        }

        // ---- C-layout P -> LDS -> A-layout fragments ----------------------
#pragma unroll
        for (int j = 0; j < 4; ++j)
#pragma unroll
            for (int r = 0; r < 8; ++r)
                Pb[w][(mo + r) * 64 + j * 16 + n] = (_Float16)st[j][r];
        asm volatile("" ::: "memory");   // keep DS store->load program order

        // ---- O += P V -----------------------------------------------------
#pragma unroll
        for (int kk = 0; kk < 2; ++kk) {
            v16h ap = load_a16x32(&Pb[w][0] + kk * 32, 64, lane);
#pragma unroll
            for (int j = 0; j < 4; ++j) {
                // B[k][n] = Vt[d = j*16+n][key]; K-contiguous in Vt.
                v16h bv = load_bT16(&Vt[(size_t)(j * 16) * 64 + kk * 32], 64, lane);
                o[j] = WMMA_F32_F16(ap, bv, o[j]);
            }
        }
        buf ^= 1;
    }

    // ---- normalize and store attn output as [B,S,D] f16 -------------------
    const int bI = bh / 12, h = bh % 12;
#pragma unroll
    for (int r = 0; r < 8; ++r) {
        float inv = 1.0f / lrow[r];
        int   qi  = qrow + mo + r;
#pragma unroll
        for (int j = 0; j < 4; ++j)
            aoh[((size_t)bI * 2048 + qi) * 768 + h * 64 + j * 16 + n] =
                (_Float16)(o[j][r] * inv);
    }
}

// ---- output projection: [8192,768] @ Wo^T -> fp32 out ---------------------
// grid: (12, 64), 128 threads; wave tile = 32x64.
__global__ void out_gemm_kernel(const _Float16* __restrict__ aoh,
                                const _Float16* __restrict__ woh,
                                float* __restrict__ out) {
    const int lane = threadIdx.x & 31;
    const int wv   = threadIdx.x >> 5;
    const int m0 = blockIdx.y * 128 + wv * 32;
    const int n0 = blockIdx.x * 64;

    v8f acc[2][4] = {};
    for (int k = 0; k < 768; k += 32) {
        v16h a0 = load_a16x32(aoh + (size_t)m0 * 768 + k,        768, lane);
        v16h a1 = load_a16x32(aoh + (size_t)(m0 + 16) * 768 + k, 768, lane);
#pragma unroll
        for (int j = 0; j < 4; ++j) {
            v16h b = load_bT16(woh + (size_t)(n0 + j * 16) * 768 + k, 768, lane);
            acc[0][j] = WMMA_F32_F16(a0, b, acc[0][j]);
            acc[1][j] = WMMA_F32_F16(a1, b, acc[1][j]);
        }
    }
    const int n  = lane & 15;
    const int mo = (lane & 16) ? 8 : 0;
#pragma unroll
    for (int hrow = 0; hrow < 2; ++hrow)
#pragma unroll
        for (int j = 0; j < 4; ++j)
#pragma unroll
            for (int r = 0; r < 8; ++r)
                out[(size_t)(m0 + hrow * 16 + mo + r) * 768 + n0 + j * 16 + n] =
                    acc[hrow][j][r];
}

// ---------------------------------------------------------------------------

extern "C" void kernel_launch(void* const* d_in, const int* in_sizes, int n_in,
                              void* d_out, int out_size, void* d_ws, size_t ws_size,
                              hipStream_t stream) {
    const float* x  = (const float*)d_in[0];
    const float* wq = (const float*)d_in[1];
    const float* wk = (const float*)d_in[2];
    const float* wv = (const float*)d_in[3];
    const float* wo = (const float*)d_in[4];

    const size_t XE = (size_t)8192 * 768;   // 6,291,456
    const size_t WE = (size_t)768 * 768;    //   589,824

    _Float16* ws  = (_Float16*)d_ws;
    _Float16* xh  = ws;             // x as f16
    _Float16* wqh = xh  + XE;       // wq*0.125, wk, wv contiguous for qkv kernel
    _Float16* wkh = wqh + WE;
    _Float16* wvh = wkh + WE;
    _Float16* woh = wvh + WE;
    _Float16* qh  = woh + WE;       // q,k,v contiguous: [3][B,H,S,64]
    _Float16* kh  = qh  + XE;
    _Float16* vh  = kh  + XE;
    _Float16* aoh = vh  + XE;       // attn output [B,S,D] f16

    cvt_f32_to_f16_kernel<<<(int)((XE + 255) / 256), 256, 0, stream>>>(x,  xh,  (int)XE, 1.0f);
    cvt_f32_to_f16_kernel<<<(int)((WE + 255) / 256), 256, 0, stream>>>(wq, wqh, (int)WE, 0.125f);
    cvt_f32_to_f16_kernel<<<(int)((WE + 255) / 256), 256, 0, stream>>>(wk, wkh, (int)WE, 1.0f);
    cvt_f32_to_f16_kernel<<<(int)((WE + 255) / 256), 256, 0, stream>>>(wv, wvh, (int)WE, 1.0f);
    cvt_f32_to_f16_kernel<<<(int)((WE + 255) / 256), 256, 0, stream>>>(wo, woh, (int)WE, 1.0f);

    qkv_gemm_kernel<<<dim3(12, 64, 3), 128, 0, stream>>>(xh, wqh, qh);
    flash_attn_kernel<<<dim3(32, 48), 128, 0, stream>>>(qh, kh, vh, aoh);
    out_gemm_kernel<<<dim3(12, 64), 128, 0, stream>>>(aoh, woh, (float*)d_out);
}
